// MultiScaleRetention_15238543966724
// MI455X (gfx1250) — matrix operand order repaired and checked by reading, not compiled, not run on hardware
//
#include <hip/hip_runtime.h>
#include <hip/hip_bf16.h>

typedef _Float16 f16;
typedef __attribute__((ext_vector_type(16))) _Float16 v16h;
typedef __attribute__((ext_vector_type(8)))  float    v8f;

#define WMMA_F16(a, b, c) \
  __builtin_amdgcn_wmma_f32_16x16x32_f16(false, (a), false, (b), (short)0, (c), false, false)

constexpr int Bc = 2, Tc = 2048, DM = 1024, NH = 16, HD = 64;

// ---------------------------------------------------------------------------
// Async global -> LDS copy (CDNA5 GLOBAL_LOAD_ASYNC_TO_LDS_B128, ASYNCcnt).
// LDS offset = low 32 bits of the flat shared-aperture address (ISA 10.2).
// ---------------------------------------------------------------------------
__device__ __forceinline__ void async_b128(f16* lds, const f16* g) {
  unsigned loff = (unsigned)(size_t)lds;
  unsigned long long ga = (unsigned long long)(size_t)g;
  asm volatile("global_load_async_to_lds_b128 %0, %1, off" ::"v"(loff), "v"(ga)
               : "memory");
}
__device__ __forceinline__ void wait_async0() {
  asm volatile("s_wait_asynccnt 0x0" ::: "memory");
}

// ---------------------------------------------------------------------------
// fp32 -> f16 conversion
// ---------------------------------------------------------------------------
__global__ void cvt_f32_f16(const float* __restrict__ src, f16* __restrict__ dst, int n) {
  int i = blockIdx.x * 256 + threadIdx.x;
  if (i < n) dst[i] = (f16)src[i];
}

// ---------------------------------------------------------------------------
// C(MxN) = A(MxK,f16) * W(NxK,f16)^T   (torch Linear: y = x @ W.T)
// 256 threads = 8 waves, block tile 128(M) x 128(N), K-step 32.
// Waves 4(M) x 2(N); each wave 32x64 = 8 accumulators, 8 WMMA/iter.
// Double-buffered LDS: async DMA of tile k+1 overlaps WMMA of tile k.
// MODE 0: f16 output scattered to (B,H,T,HD), scaled.  MODE 1: f32 row-major.
// ---------------------------------------------------------------------------
template <int MODE>
__global__ __launch_bounds__(256) void gemm_wmma(const f16* __restrict__ A,
                                                 const f16* __restrict__ W,
                                                 f16* __restrict__ outh,
                                                 float* __restrict__ outf,
                                                 float scale) {
  constexpr int Kdim = DM;
  constexpr int LS = 40;  // LDS row stride (32 + pad) in halves
  __shared__ f16 Al[2][128 * LS];
  __shared__ f16 Wl[2][128 * LS];

  const int tid   = threadIdx.x;
  const int lane  = tid & 31;
  const int wv    = tid >> 5;
  const int lo16  = lane & 15;
  const int half8 = (lane & 16) ? 8 : 0;
  const int wm = (wv & 3) * 32;   // wave M base in tile
  const int wn = (wv >> 2) * 64;  // wave N base in tile
  const int m0 = blockIdx.x * 128;
  const int n0 = blockIdx.y * 128;

  auto load_tiles = [&](int k0, int buf) {
#pragma unroll
    for (int i = 0; i < 2; ++i) {
      int c = tid + i * 256;
      int r = c >> 2, q = (c & 3) * 8;
      async_b128(&Al[buf][r * LS + q], &A[(size_t)(m0 + r) * Kdim + k0 + q]);
      async_b128(&Wl[buf][r * LS + q], &W[(size_t)(n0 + r) * Kdim + k0 + q]);
    }
  };

  v8f acc[2][4] = {};
  load_tiles(0, 0);

  int buf = 0;
  for (int k0 = 0; k0 < Kdim; k0 += 32, buf ^= 1) {
    wait_async0();
    __syncthreads();  // buf complete everywhere; prior reads of buf^1 done
    if (k0 + 32 < Kdim) load_tiles(k0 + 32, buf ^ 1);  // DMA overlaps compute

    // A fragments (ISA 16-bit A layout: two 16B chunks at K = kh, kh+16)
    const int kh = half8;
    v16h af[2];
#pragma unroll
    for (int mt = 0; mt < 2; ++mt) {
      const f16* ap = &Al[buf][(wm + mt * 16 + lo16) * LS + kh];
      ((uint4*)&af[mt])[0] = *(const uint4*)(ap);
      ((uint4*)&af[mt])[1] = *(const uint4*)(ap + 16);
    }
    const int kb = (lane & 16) ? 8 : 0;
#pragma unroll
    for (int nt = 0; nt < 4; ++nt) {
      v16h bf;  // B fragment: two 8-half K chunks of row n at kb, kb+16
      const f16* bp = &Wl[buf][(wn + nt * 16 + lo16) * LS + kb];
      ((uint4*)&bf)[0] = *(const uint4*)(bp);
      ((uint4*)&bf)[1] = *(const uint4*)(bp + 16);
#pragma unroll
      for (int mt = 0; mt < 2; ++mt) acc[mt][nt] = WMMA_F16(af[mt], bf, acc[mt][nt]);
    }
  }

  // Store: D layout — lane holds N = lo16, VGPR r holds M = r + half8
#pragma unroll
  for (int nt = 0; nt < 4; ++nt) {
    const int n = n0 + wn + nt * 16 + lo16;
#pragma unroll
    for (int mt = 0; mt < 2; ++mt) {
#pragma unroll
      for (int r = 0; r < 8; ++r) {
        const int m = m0 + wm + mt * 16 + r + half8;
        if (MODE == 0) {
          const int hh = n >> 6, d = n & 63;
          const int bb = m >> 11, t = m & 2047;  // T = 2048
          outh[(((size_t)bb * NH + hh) * Tc + t) * HD + d] = (f16)(acc[mt][nt][r] * scale);
        } else {
          __builtin_nontemporal_store(acc[mt][nt][r], &outf[(size_t)m * DM + n]);
        }
      }
    }
  }
}

// ---------------------------------------------------------------------------
// Retention: per (b,h):  out = (Q K^T * decayMask) V,  Q pre-scaled by 1/8.
// Grid: (T/256, B*H). 8 waves/block, each wave owns 32 query rows.
// Double-buffered K/V tiles; decay mask via 1 exp2 + cumulative gamma-multiply.
// ---------------------------------------------------------------------------
__global__ __launch_bounds__(256) void retention_wmma(const f16* __restrict__ Q,
                                                      const f16* __restrict__ K,
                                                      const f16* __restrict__ V,
                                                      float* __restrict__ Y) {
  constexpr int KS = 72;  // K tile row stride (64 + pad)
  constexpr int VS = 40;  // Vt row stride (32 + pad)
  constexpr int SS = 40;  // S scratch row stride
  __shared__ f16 Kl[2][32 * KS];
  __shared__ f16 Vt[2][64 * VS];   // V tile transposed: Vt[d][j]
  __shared__ f16 Sl[8 * 32 * SS];  // per-wave S scratch (32 rows)

  const int tid   = threadIdx.x;
  const int lane  = tid & 31;
  const int wv    = tid >> 5;
  const int lo16  = lane & 15;
  const int half8 = (lane & 16) ? 8 : 0;
  const int bh  = blockIdx.y;
  const int h   = bh & (NH - 1);
  const int i0b = blockIdx.x * 256;
  const int i0  = i0b + wv * 32;

  const f16* Qp = Q + (size_t)bh * Tc * HD;
  const f16* Kp = K + (size_t)bh * Tc * HD;
  const f16* Vp = V + (size_t)bh * Tc * HD;
  const float gam   = 1.0f - exp2f(-5.0f - (float)h * 0.5f);
  const float log2g = log2f(gam);

  auto load_kv = [&](int jc, int b) {
    {  // K tile 32x64 via async DMA into LDS
      int r = tid >> 3, q = (tid & 7) * 8;
      async_b128(&Kl[b][r * KS + q], &Kp[(size_t)(jc + r) * HD + q]);
    }
    {  // V tile transposed into Vt[d][j]
      int j = tid & 31, db = (tid >> 5) * 8;
      uint4 vv = *(const uint4*)&Vp[(size_t)(jc + j) * HD + db];
      const f16* vp = (const f16*)&vv;
#pragma unroll
      for (int q = 0; q < 8; ++q) Vt[b][(db + q) * VS + j] = vp[q];
    }
  };

  // Q fragments for 2 M-subtiles x 2 d-chunks, resident in VGPRs for the pass
  const int kh = half8;
  v16h qf[2][2];
#pragma unroll
  for (int mt = 0; mt < 2; ++mt) {
    const f16* qp = &Qp[(size_t)(i0 + mt * 16 + lo16) * HD + kh];
    ((uint4*)&qf[mt][0])[0] = *(const uint4*)(qp);
    ((uint4*)&qf[mt][0])[1] = *(const uint4*)(qp + 16);
    ((uint4*)&qf[mt][1])[0] = *(const uint4*)(qp + 32);
    ((uint4*)&qf[mt][1])[1] = *(const uint4*)(qp + 48);
  }

  v8f acc[2][4] = {};
  const int jmax = i0b + 256;  // causal bound for the whole block
  load_kv(0, 0);

  int buf = 0;
  for (int jc = 0; jc < jmax; jc += 32, buf ^= 1) {
    wait_async0();
    __syncthreads();  // buf K/V complete; prior reads of buf^1 done
    if (jc + 32 < jmax) load_kv(jc + 32, buf ^ 1);  // DMA overlaps compute

    // S = Q K^T (masked) for 2 j-subtiles x 2 m-subtiles
    f16* sw = &Sl[wv * 32 * SS];
    const int kb = (lane & 16) ? 8 : 0;
#pragma unroll
    for (int jt = 0; jt < 2; ++jt) {
      v16h b0, b1;
      const f16* brow = &Kl[buf][(jt * 16 + lo16) * KS];
      ((uint4*)&b0)[0] = *(const uint4*)(brow + kb);
      ((uint4*)&b0)[1] = *(const uint4*)(brow + kb + 16);
      ((uint4*)&b1)[0] = *(const uint4*)(brow + 32 + kb);
      ((uint4*)&b1)[1] = *(const uint4*)(brow + 32 + kb + 16);
      const int j0 = jc + jt * 16 + lo16;
#pragma unroll
      for (int mt = 0; mt < 2; ++mt) {
        v8f s = {};
        s = WMMA_F16(qf[mt][0], b0, s);
        s = WMMA_F16(qf[mt][1], b1, s);
        // decay mask: gamma^(i-j) for i>=j else 0; cumulative along r
        const int df0 = i0 + mt * 16 + half8 - j0;
        float gcur = __builtin_amdgcn_exp2f((float)df0 * log2g);
#pragma unroll
        for (int r = 0; r < 8; ++r) {
          const float dm = (df0 + r >= 0) ? gcur : 0.0f;
          sw[(mt * 16 + r + half8) * SS + jt * 16 + lo16] = (f16)(s[r] * dm);
          gcur *= gam;
        }
      }
    }
    // per-wave LDS permute (D-layout -> A-layout): drain DS before re-reading
    asm volatile("s_wait_dscnt 0x0" ::: "memory");

    v16h sa[2];
#pragma unroll
    for (int mt = 0; mt < 2; ++mt) {
      const f16* sp = &sw[(mt * 16 + lo16) * SS + kh];
      ((uint4*)&sa[mt])[0] = *(const uint4*)(sp);
      ((uint4*)&sa[mt])[1] = *(const uint4*)(sp + 16);
    }
    const int kb2 = (lane & 16) ? 8 : 0;
#pragma unroll
    for (int nt = 0; nt < 4; ++nt) {
      v16h bv;  // V^T fragment: n = d, contiguous over j
      const f16* vp2 = &Vt[buf][(nt * 16 + lo16) * VS + kb2];
      ((uint4*)&bv)[0] = *(const uint4*)(vp2);
      ((uint4*)&bv)[1] = *(const uint4*)(vp2 + 16);
#pragma unroll
      for (int mt = 0; mt < 2; ++mt) acc[mt][nt] = WMMA_F16(sa[mt], bv, acc[mt][nt]);
    }
  }

  // write fp32 retention output in (B,H,T,D) for GroupNorm
  float* Yp = Y + (size_t)bh * Tc * HD;
#pragma unroll
  for (int nt = 0; nt < 4; ++nt) {
    const int d = nt * 16 + lo16;
#pragma unroll
    for (int mt = 0; mt < 2; ++mt) {
#pragma unroll
      for (int r = 0; r < 8; ++r) {
        const int i = i0 + mt * 16 + r + half8;
        Yp[(size_t)i * HD + d] = acc[mt][nt][r];
      }
    }
  }
}

// ---------------------------------------------------------------------------
// GroupNorm over (D, T) per (b,h); writes normalized f16 in (B,T,C) layout.
// ---------------------------------------------------------------------------
__global__ __launch_bounds__(256) void groupnorm_k(const float* __restrict__ Y,
                                                   const float* __restrict__ gw,
                                                   const float* __restrict__ gb,
                                                   f16* __restrict__ Z) {
  __shared__ float ssum[256], ssq[256];
  const int bh = blockIdx.x;
  const int b = bh >> 4, h = bh & 15;
  const float* Yp = Y + (size_t)bh * Tc * HD;
  const int N = Tc * HD;  // 131072

  float s = 0.f, sq = 0.f;
  const float4* Y4 = (const float4*)Yp;
  for (int i = threadIdx.x; i < N / 4; i += 256) {
    float4 v = Y4[i];
    s  += v.x + v.y + v.z + v.w;
    sq += v.x * v.x + v.y * v.y + v.z * v.z + v.w * v.w;
  }
  ssum[threadIdx.x] = s;
  ssq[threadIdx.x] = sq;
  __syncthreads();
  for (int st = 128; st > 0; st >>= 1) {
    if (threadIdx.x < st) {
      ssum[threadIdx.x] += ssum[threadIdx.x + st];
      ssq[threadIdx.x] += ssq[threadIdx.x + st];
    }
    __syncthreads();
  }
  const float mu  = ssum[0] / (float)N;
  const float var = ssq[0] / (float)N - mu * mu;
  const float rs  = rsqrtf(var + 1e-5f);

  for (int i = threadIdx.x * 2; i < N; i += 512) {
    const int t = i >> 6, d = i & 63;  // d even
    const int c = h * HD + d;
    const float z0 = (Yp[i] - mu) * rs * gw[c] + gb[c];
    const float z1 = (Yp[i + 1] - mu) * rs * gw[c + 1] + gb[c + 1];
    union { f16 hh[2]; unsigned u; } pk;
    pk.hh[0] = (f16)z0;
    pk.hh[1] = (f16)z1;
    *(unsigned*)&Z[(size_t)(b * Tc + t) * DM + c] = pk.u;
  }
}

// ---------------------------------------------------------------------------
extern "C" void kernel_launch(void* const* d_in, const int* in_sizes, int n_in,
                              void* d_out, int out_size, void* d_ws, size_t ws_size,
                              hipStream_t stream) {
  const float* x  = (const float*)d_in[0];
  const float* Wq = (const float*)d_in[1];
  const float* Wk = (const float*)d_in[2];
  const float* Wv = (const float*)d_in[3];
  const float* Wo = (const float*)d_in[4];
  const float* gw = (const float*)d_in[5];
  const float* gb = (const float*)d_in[6];

  const size_t SZX = (size_t)Bc * Tc * DM;  // 4096*1024
  const size_t SZW = (size_t)DM * DM;

  char* p = (char*)d_ws;
  f16* xh  = (f16*)p;    p += SZX * 2;
  f16* wqh = (f16*)p;    p += SZW * 2;
  f16* wkh = (f16*)p;    p += SZW * 2;
  f16* wvh = (f16*)p;    p += SZW * 2;
  f16* woh = (f16*)p;    p += SZW * 2;
  f16* qh  = (f16*)p;    p += SZX * 2;
  f16* kh  = (f16*)p;    p += SZX * 2;
  f16* vh  = (f16*)p;    p += SZX * 2;
  float* yf = (float*)p; p += SZX * 4;
  f16* zh  = (f16*)p;    p += SZX * 2;

  cvt_f32_f16<<<(int)((SZX + 255) / 256), 256, 0, stream>>>(x, xh, (int)SZX);
  cvt_f32_f16<<<(int)((SZW + 255) / 256), 256, 0, stream>>>(Wq, wqh, (int)SZW);
  cvt_f32_f16<<<(int)((SZW + 255) / 256), 256, 0, stream>>>(Wk, wkh, (int)SZW);
  cvt_f32_f16<<<(int)((SZW + 255) / 256), 256, 0, stream>>>(Wv, wvh, (int)SZW);
  cvt_f32_f16<<<(int)((SZW + 255) / 256), 256, 0, stream>>>(Wo, woh, (int)SZW);

  dim3 gg(Bc * Tc / 128, DM / 128);  // 32 x 8
  gemm_wmma<0><<<gg, 256, 0, stream>>>(xh, wqh, qh, nullptr, 0.125f);  // Q / sqrt(D)
  gemm_wmma<0><<<gg, 256, 0, stream>>>(xh, wkh, kh, nullptr, 1.0f);
  gemm_wmma<0><<<gg, 256, 0, stream>>>(xh, wvh, vh, nullptr, 1.0f);

  dim3 gr(Tc / 256, Bc * NH);  // 8 x 32
  retention_wmma<<<gr, 256, 0, stream>>>(qh, kh, vh, yf);

  groupnorm_k<<<Bc * NH, 256, 0, stream>>>(yf, gw, gb, zh);

  gemm_wmma<1><<<gg, 256, 0, stream>>>(zh, woh, nullptr, (float*)d_out, 1.0f);
}
